// FC_LSTM_59064390255071
// MI455X (gfx1250) — matrix operand order repaired and checked by reading, not compiled, not run on hardware
//
#include <hip/hip_runtime.h>
#include <hip/hip_bf16.h>

typedef __attribute__((ext_vector_type(16))) __bf16 v16bf;
typedef __attribute__((ext_vector_type(8)))  __bf16 v8bf;
typedef __attribute__((ext_vector_type(8)))  float  v8f;

__device__ __forceinline__ float sigmoidf_(float x) {
    return 1.0f / (1.0f + __expf(-x));
}

__device__ __forceinline__ v8f wmma_bf16(v16bf a, v16bf b, v8f c) {
    // (neg_a, A, neg_b, B, c_mod, C, reuse_a, reuse_b)
    return __builtin_amdgcn_wmma_f32_16x16x32_bf16(false, a, false, b, (short)0, c, false, false);
}

// A fragment (16x32 bf16): lane m+16g holds row m, contiguous K chunk [k0+16g,+16).
__device__ __forceinline__ v16bf load_a_frag_bf(const __bf16* __restrict__ xrow, int k0, int g) {
    return *(const v16bf*)(xrow + k0 + 16 * g);
}

// B fragment (32x16 bf16): lane n+16g holds W row n, contiguous K chunk [k0+16g,+16).
__device__ __forceinline__ v16bf load_b_frag_bf(const __bf16* __restrict__ wrow, int k0, int g) {
    return *(const v16bf*)(wrow + k0 + 16 * g);
}

// ---------------- convert & zero fill ----------------

__global__ __launch_bounds__(256) void f32_to_bf16_kernel(const float* __restrict__ src,
                                                          __bf16* __restrict__ dst, int n) {
    for (int i = blockIdx.x * blockDim.x + threadIdx.x; i < n; i += gridDim.x * blockDim.x)
        dst[i] = (__bf16)src[i];
}

__global__ __launch_bounds__(256) void zero_u32_kernel(unsigned int* __restrict__ dst, int n) {
    for (int i = blockIdx.x * blockDim.x + threadIdx.x; i < n; i += gridDim.x * blockDim.x)
        dst[i] = 0u;
}

// ---------------- fused GEMM + bias + activation ----------------
// Y(256,N) = act(X(256,K)_bf16 @ W(N,K)_bf16^T + b)
// One wave per 32x64 macro-tile: 2 M-tiles (rows m0, m0+128) x 4 N-tiles,
// 8 accumulators (64 VGPRs) -- forces the allocator past the 64-reg heuristic
// and amortizes each B fragment over 2 WMMAs.
// ACT: 1 = relu, 2 = tanh.  OBF: bf16 vs f32 stores.

template <int ACT, int N, int K, bool OBF>
__global__ __launch_bounds__(256, 1) __attribute__((amdgpu_waves_per_eu(1)))
void gemm_bias_act(const __bf16* __restrict__ X,
                   const __bf16* __restrict__ W,
                   const float* __restrict__ bias,
                   void* __restrict__ Yv) {
    constexpr int NT4 = N / 64;
    const int lane = threadIdx.x & 31;
    const int wid  = blockIdx.x * 8 + (threadIdx.x >> 5);   // 8*NT4 waves total
    const int m0   = (wid / NT4) << 4;                      // rows m0.. and m0+128..
    const int n0   = (wid % NT4) << 6;
    const int g    = lane >> 4;
    const int l15  = lane & 15;

    const __bf16* xrow0 = X + (size_t)(m0 + l15) * K;
    const __bf16* xrow1 = X + (size_t)(m0 + 128 + l15) * K;
    const __bf16* wbase = W + (size_t)(n0 + l15) * K;

    v8f acc[8] = {{}, {}, {}, {}, {}, {}, {}, {}};

#pragma unroll 2
    for (int k0 = 0; k0 < K; k0 += 32) {
        v16bf a0 = load_a_frag_bf(xrow0, k0, g);
        v16bf a1 = load_a_frag_bf(xrow1, k0, g);
#pragma unroll
        for (int j = 0; j < 4; ++j) {
            v16bf b = load_b_frag_bf(wbase + (size_t)j * 16 * K, k0, g);
            acc[j]     = wmma_bf16(a0, b, acc[j]);
            acc[4 + j] = wmma_bf16(a1, b, acc[4 + j]);
        }
    }

#pragma unroll
    for (int j = 0; j < 4; ++j) {
        const int   col = n0 + j * 16 + l15;
        const float bv  = bias[col];
#pragma unroll
        for (int h = 0; h < 2; ++h) {
            const int rbase = m0 + h * 128;
#pragma unroll
            for (int r = 0; r < 8; ++r) {
                float v = acc[h * 4 + j][r] + bv;
                if (ACT == 1) v = fmaxf(v, 0.0f);
                if (ACT == 2) v = tanhf(v);
                const size_t idx = (size_t)(rbase + r + 8 * g) * N + col;
                if (OBF) ((__bf16*)Yv)[idx] = (__bf16)v;
                else     ((float*)Yv)[idx]  = v;
            }
        }
    }
}

// ---------------- fused LSTM cell ----------------
// gates = X @ Wih^T + Hin @ Whh^T + bih + bhh  (X,Hin: 256x256 bf16; W: 1024x256 bf16)
// One wave owns a 32x16 (batch, hidden) patch across all 4 gates: 2 M-tiles
// (rows m0, m0+128) x 4 gate accumulators = 8 accumulators; each gate's B
// fragment is reused by both M-tiles.  C fp32 in-place (exclusive per-wave
// ownership); H ping-ponged bf16.

__global__ __launch_bounds__(256, 1) __attribute__((amdgpu_waves_per_eu(1)))
void lstm_cell_kernel(const __bf16* __restrict__ X,
                      const __bf16* __restrict__ Hin,
                      const __bf16* __restrict__ Wih,
                      const __bf16* __restrict__ Whh,
                      const float* __restrict__ bih,
                      const float* __restrict__ bhh,
                      float* __restrict__ C,
                      __bf16* __restrict__ Hout) {
    const int lane = threadIdx.x & 31;
    const int wid  = blockIdx.x * 8 + (threadIdx.x >> 5);   // 128 waves total
    const int m0   = (wid >> 4) << 4;                       // 8 M macro-tiles
    const int n0   = (wid & 15) << 4;                       // 16 hidden tiles
    const int g    = lane >> 4;
    const int l15  = lane & 15;

    const __bf16* xrow0 = X   + (size_t)(m0 + l15) * 256;
    const __bf16* xrow1 = X   + (size_t)(m0 + 128 + l15) * 256;
    const __bf16* hrow0 = Hin + (size_t)(m0 + l15) * 256;
    const __bf16* hrow1 = Hin + (size_t)(m0 + 128 + l15) * 256;
    const __bf16* wi    = Wih + (size_t)(n0 + l15) * 256;
    const __bf16* wh    = Whh + (size_t)(n0 + l15) * 256;

    v8f acc[8] = {{}, {}, {}, {}, {}, {}, {}, {}};

#pragma unroll 2
    for (int k0 = 0; k0 < 256; k0 += 32) {        // X @ Wih^T
        v16bf a0 = load_a_frag_bf(xrow0, k0, g);
        v16bf a1 = load_a_frag_bf(xrow1, k0, g);
#pragma unroll
        for (int j = 0; j < 4; ++j) {
            v16bf b = load_b_frag_bf(wi + (size_t)j * 256 * 256, k0, g);
            acc[j]     = wmma_bf16(a0, b, acc[j]);
            acc[4 + j] = wmma_bf16(a1, b, acc[4 + j]);
        }
    }
#pragma unroll 2
    for (int k0 = 0; k0 < 256; k0 += 32) {        // Hin @ Whh^T
        v16bf a0 = load_a_frag_bf(hrow0, k0, g);
        v16bf a1 = load_a_frag_bf(hrow1, k0, g);
#pragma unroll
        for (int j = 0; j < 4; ++j) {
            v16bf b = load_b_frag_bf(wh + (size_t)j * 256 * 256, k0, g);
            acc[j]     = wmma_bf16(a0, b, acc[j]);
            acc[4 + j] = wmma_bf16(a1, b, acc[4 + j]);
        }
    }

    const int   col = n0 + l15;
    const float b0  = bih[col]       + bhh[col];
    const float b1  = bih[256 + col] + bhh[256 + col];
    const float b2  = bih[512 + col] + bhh[512 + col];
    const float b3  = bih[768 + col] + bhh[768 + col];

#pragma unroll
    for (int h = 0; h < 2; ++h) {
        const int rbase = m0 + h * 128;
#pragma unroll
        for (int r = 0; r < 8; ++r) {
            const size_t idx = (size_t)(rbase + r + 8 * g) * 256 + col;
            float iv = sigmoidf_(acc[h * 4 + 0][r] + b0);
            float fv = sigmoidf_(acc[h * 4 + 1][r] + b1);
            float gv = tanhf   (acc[h * 4 + 2][r] + b2);
            float ov = sigmoidf_(acc[h * 4 + 3][r] + b3);
            float cn = fv * C[idx] + iv * gv;
            C[idx]    = cn;
            Hout[idx] = (__bf16)(ov * tanhf(cn));
        }
    }
}

// ---------------- host orchestration ----------------

extern "C" void kernel_launch(void* const* d_in, const int* in_sizes, int n_in,
                              void* d_out, int out_size, void* d_ws, size_t ws_size,
                              hipStream_t stream) {
    const float* x        = (const float*)d_in[0];
    const float* fc_en1_w = (const float*)d_in[1];
    const float* fc_en1_b = (const float*)d_in[2];
    const float* fc_en2_w = (const float*)d_in[3];
    const float* fc_en2_b = (const float*)d_in[4];
    // cells: c=0..5 -> wih=5+4c, whh=6+4c, bih=7+4c, bhh=8+4c
    const float* fc_de1_w = (const float*)d_in[29];
    const float* fc_de1_b = (const float*)d_in[30];
    const float* fc_de2_w = (const float*)d_in[31];
    const float* fc_de2_b = (const float*)d_in[32];

    const int fut = out_size / (256 * 64 * 64);   // future_step derived host-side
    float* out = (float*)d_out;

    // --- workspace carve-up ---
    char*  base = (char*)d_ws;
    size_t off  = 0;
    auto carve = [&](size_t bytes) { void* p = base + off; off += (bytes + 255) & ~(size_t)255; return p; };
    auto abf   = [&](size_t n) { return (__bf16*)carve(n * 2); };

    __bf16* wb_en1 = abf((size_t)1024 * 4096);
    __bf16* wb_en2 = abf((size_t)256 * 1024);
    __bf16* wih[6];
    __bf16* whh[6];
    for (int c = 0; c < 6; ++c) {
        wih[c] = abf((size_t)1024 * 256);
        whh[c] = abf((size_t)1024 * 256);
    }
    __bf16* wb_de1 = abf((size_t)1024 * 256);
    __bf16* wb_de2 = abf((size_t)4096 * 1024);

    __bf16* xb = abf((size_t)256 * 4096);   // per-step bf16 copy of x slice
    __bf16* z1 = abf((size_t)256 * 1024);
    __bf16* z2 = abf((size_t)256 * 256);
    __bf16* y1 = abf((size_t)256 * 1024);

    // contiguous zero-init block: bf16 [zero_x | 12 h buffers] then f32 [6 c buffers]
    const size_t zbytes = (size_t)13 * 65536 * 2 + (size_t)6 * 65536 * 4;
    char* zblock = (char*)carve(zbytes);
    __bf16* zero_x = (__bf16*)zblock;
    __bf16* hbuf[6][2];
    float*  cbuf[6];
    for (int c = 0; c < 6; ++c) {
        hbuf[c][0] = (__bf16*)zblock + (size_t)65536 * (1 + 2 * c);
        hbuf[c][1] = (__bf16*)zblock + (size_t)65536 * (2 + 2 * c);
        cbuf[c]    = (float*)(zblock + (size_t)13 * 65536 * 2) + (size_t)65536 * c;
    }

    // --- weight conversion fp32 -> bf16 (~24 MB, L2-resident afterwards) ---
    auto conv = [&](const float* s, __bf16* d, int n) {
        f32_to_bf16_kernel<<<512, 256, 0, stream>>>(s, d, n);
    };
    conv(fc_en1_w, wb_en1, 1024 * 4096);
    conv(fc_en2_w, wb_en2, 256 * 1024);
    for (int c = 0; c < 6; ++c) {
        conv((const float*)d_in[5 + 4 * c], wih[c], 1024 * 256);
        conv((const float*)d_in[6 + 4 * c], whh[c], 1024 * 256);
    }
    conv(fc_de1_w, wb_de1, 1024 * 256);
    conv(fc_de2_w, wb_de2, 4096 * 1024);

    zero_u32_kernel<<<512, 256, 0, stream>>>((unsigned int*)zblock, (int)(zbytes / 4));

    // --- encoder: 20 sequential steps ---
    for (int t = 0; t < 20; ++t) {
        f32_to_bf16_kernel<<<1024, 256, 0, stream>>>(x + (size_t)t * 256 * 4096, xb, 256 * 4096);
        gemm_bias_act<1, 1024, 4096, true><<<16, 256, 0, stream>>>(xb, wb_en1, fc_en1_b, z1);
        gemm_bias_act<1, 256, 1024, true><<<4, 256, 0, stream>>>(z1, wb_en2, fc_en2_b, z2);
        const __bf16* xi = z2;
        for (int c = 0; c < 3; ++c) {
            lstm_cell_kernel<<<16, 256, 0, stream>>>(xi, hbuf[c][t & 1], wih[c], whh[c],
                                                     (const float*)d_in[7 + 4 * c],
                                                     (const float*)d_in[8 + 4 * c],
                                                     cbuf[c], hbuf[c][(t + 1) & 1]);
            xi = hbuf[c][(t + 1) & 1];
        }
    }

    // decoder h1 initial state = final encoder h3 (20 steps -> ping index 0)
    hipMemcpyAsync(hbuf[3][0], hbuf[2][0], (size_t)65536 * sizeof(__bf16),
                   hipMemcpyDeviceToDevice, stream);

    // --- decoder: fut sequential steps ---
    for (int t = 0; t < fut; ++t) {
        const __bf16* xi = zero_x;
        for (int c = 3; c < 6; ++c) {
            lstm_cell_kernel<<<16, 256, 0, stream>>>(xi, hbuf[c][t & 1], wih[c], whh[c],
                                                     (const float*)d_in[7 + 4 * c],
                                                     (const float*)d_in[8 + 4 * c],
                                                     cbuf[c], hbuf[c][(t + 1) & 1]);
            xi = hbuf[c][(t + 1) & 1];
        }
        gemm_bias_act<1, 1024, 256, true><<<16, 256, 0, stream>>>(xi, wb_de1, fc_de1_b, y1);
        gemm_bias_act<2, 4096, 1024, false><<<64, 256, 0, stream>>>(y1, wb_de2, fc_de2_b,
                                                                    out + (size_t)t * 256 * 4096);
    }
}
